// AGRNN_44023414784183
// MI455X (gfx1250) — matrix operand order repaired and checked by reading, not compiled, not run on hardware
//
#include <hip/hip_runtime.h>
#include <hip/hip_bf16.h>

// ---------------------------------------------------------------------------
// AGRNN edge classifier on MI455X (gfx1250).
// All four GEMMs run on v_wmma_f32_16x16x32_bf16 (bf16 A/B, f32 accumulate).
// Weights are pre-transposed/converted to padded bf16 once, so every tile
// load is a 16-byte vector op; A-side gather-concat is fused into chunked
// 16-element vector loads with a scalar fallback only at segment seams.
// ---------------------------------------------------------------------------

typedef __bf16 bf16_t;
typedef __attribute__((ext_vector_type(16))) __bf16 v16bf;
typedef __attribute__((ext_vector_type(8)))  __bf16 v8bf;
typedef __attribute__((ext_vector_type(4)))  __bf16 v4bf;
typedef __attribute__((ext_vector_type(8)))  float  v8f;
typedef __attribute__((ext_vector_type(4)))  float  v4f;

#define N_NODES 20000
#define N_EDGES 100000
#define D_FEAT  1024
#define D_LANG  300
#define NL_LD   304        // padded lang dim
#define D_SP    16
#define D_HID   1024
#define N_CLS   117
#define K_EDGE  2664       // 1024 + 300 + 16 + 300 + 1024

#define BM   128
#define BN   128
#define KT   32
#define LPAD 8             // LDS row stride 40 bf16 = 80B (16B aligned)

// Padded (rows x ldk) geometry of the pre-transposed bf16 weights
#define LDK_NODE 2048
#define LDK_LANG 608       // 600 -> 19 K-tiles
#define LDK_E1   2688      // 2664 -> 84 K-tiles
#define LDK_E2   1024
#define NR_NODE  1024
#define NR_LANG  384       // 304 -> 3 N-blocks of 128
#define NR_E1    1024
#define NR_E2    128

// ---------------- 16-element bf16 chunk helpers ----------------------------

struct Chunk { v8bf lo, hi; };

__device__ inline Chunk ld16bf(const bf16_t* p) {          // 16B aligned
    Chunk c; c.lo = *(const v8bf*)p; c.hi = *(const v8bf*)(p + 8); return c;
}
__device__ inline Chunk ld16bf_a8(const bf16_t* p) {       // 8B aligned
    Chunk c;
    v4bf a = *(const v4bf*)p,       b = *(const v4bf*)(p + 4);
    v4bf d = *(const v4bf*)(p + 8), e = *(const v4bf*)(p + 12);
#pragma unroll
    for (int j = 0; j < 4; ++j) {
        c.lo[j] = a[j]; c.lo[4 + j] = b[j];
        c.hi[j] = d[j]; c.hi[4 + j] = e[j];
    }
    return c;
}
__device__ inline Chunk cvt16(const float* p) {            // 16 fp32 -> bf16
    v4f f0 = *(const v4f*)p,       f1 = *(const v4f*)(p + 4);
    v4f f2 = *(const v4f*)(p + 8), f3 = *(const v4f*)(p + 12);
    Chunk c;
#pragma unroll
    for (int j = 0; j < 4; ++j) {
        c.lo[j] = (bf16_t)f0[j]; c.lo[4 + j] = (bf16_t)f1[j];
        c.hi[j] = (bf16_t)f2[j]; c.hi[4 + j] = (bf16_t)f3[j];
    }
    return c;
}

// ---------------- A-fetch functors (16-chunk loads, kb % 16 == 0) ----------

struct ConcatA {                 // A = [x0 | x1], fp32 -> bf16
    const float* x0; const float* x1;
    int rows, k0, ktot;
    __device__ Chunk load16(int r, int kb) const {
        Chunk c{};
        if (r >= rows || kb >= ktot) return c;
        if (kb + 16 <= k0)
            return cvt16(x0 + (long long)r * k0 + kb);
        if (kb >= k0 && kb + 16 <= ktot)
            return cvt16(x1 + (long long)r * (ktot - k0) + (kb - k0));
        const int k1 = ktot - k0;
#pragma unroll
        for (int j = 0; j < 16; ++j) {                      // seam fallback
            int k = kb + j;
            float v = (k >= ktot) ? 0.f
                    : (k < k0)    ? x0[(long long)r * k0 + k]
                                  : x1[(long long)r * k1 + (k - k0)];
            bf16_t b = (bf16_t)v;
            if (j < 8) c.lo[j] = b; else c.hi[j - 8] = b;
        }
        return c;
    }
};

struct EdgeA {                   // fused gather-concat for edge rows
    const bf16_t* nf; const bf16_t* nl; const float* sf;
    const int* src; const int* dst; int rows;
    __device__ bf16_t get1(int e, int k) const {
        if (k < 1024)   return nf[(long long)dst[e] * D_FEAT + k];
        if (k < 1324)   return nl[(long long)dst[e] * NL_LD + (k - 1024)];
        if (k < 1340)   return (bf16_t)sf[(long long)e * D_SP + (k - 1324)];
        if (k < 1640)   return nl[(long long)src[e] * NL_LD + (k - 1340)];
        if (k < K_EDGE) return nf[(long long)src[e] * D_FEAT + (k - 1640)];
        return (bf16_t)0.f;
    }
    __device__ Chunk load16(int e, int kb) const {
        Chunk c{};
        if (e >= rows) return c;
        if (kb + 16 <= 1024)
            return ld16bf(nf + (long long)dst[e] * D_FEAT + kb);
        if (kb >= 1024 && kb + 16 <= 1324)
            return ld16bf_a8(nl + (long long)dst[e] * NL_LD + (kb - 1024));
        if (kb >= 1340 && kb + 16 <= 1640)
            return ld16bf_a8(nl + (long long)src[e] * NL_LD + (kb - 1340));
        if (kb >= 1640 && kb + 16 <= K_EDGE)
            return ld16bf(nf + (long long)src[e] * D_FEAT + (kb - 1640));
#pragma unroll
        for (int j = 0; j < 16; ++j) {                      // seam fallback
            bf16_t b = get1(e, kb + j);
            if (j < 8) c.lo[j] = b; else c.hi[j - 8] = b;
        }
        return c;
    }
};

struct Bf16A {                   // A already bf16 in workspace (K % 16 == 0)
    const bf16_t* a; int rows, ld, K;
    __device__ Chunk load16(int r, int kb) const {
        Chunk c{};
        if (r >= rows || kb >= K) return c;
        return ld16bf(a + (long long)r * ld + kb);
    }
};

// ---------------- B-fetch: pre-transposed padded bf16 (branchless) ---------

struct TransB {
    const bf16_t* wt; int ldk;   // wt[n][k], rows/ldk padded to tile grid
    __device__ Chunk load16(int n, int kb) const {
        return ld16bf(wt + (long long)n * ldk + kb);
    }
};

// ---------------- Epilogue functors ----------------------------------------

struct ReluBf16Out {             // bias + ReLU -> bf16 ws (zero pad cols)
    bf16_t* out; const float* bias; int rows, cols, ld;
    __device__ void store(int r, int c, float v) const {
        if (r < rows && c < ld) {
            float x = (c < cols) ? v + bias[c] : 0.f;
            out[(long long)r * ld + c] = (bf16_t)(x > 0.f ? x : 0.f);
        }
    }
};

struct BiasF32Out {              // bias, no activation -> fp32 logits
    float* out; const float* bias; int rows, cols;
    __device__ void store(int r, int c, float v) const {
        if (r < rows && c < cols)
            out[(long long)r * cols + c] = v + bias[c];
    }
};

// ---------------- Weight transpose + bf16 convert (one-time) ---------------

__global__ __launch_bounds__(256)
void transpose_w_kernel(const float* __restrict__ w, bf16_t* __restrict__ wt,
                        int K, int N, int ldk, long long total)
{
    long long i = (long long)blockIdx.x * blockDim.x + threadIdx.x;
    if (i >= total) return;
    int n = (int)(i / ldk), k = (int)(i % ldk);
    wt[i] = (n < N && k < K) ? (bf16_t)w[(long long)k * N + n] : (bf16_t)0.f;
}

// ---------------- Generic 128x128 WMMA GEMM --------------------------------
// 256 threads = 8 waves (wave32); wave grid 4(M) x 2(N); each wave owns a
// 32x64 tile = 2x4 wmma accumulators. Next K-tile chunks are prefetched into
// registers so global traffic overlaps the WMMA burst.

template <class AF, class BF, class EP>
__global__ __launch_bounds__(256)
void wmma_gemm_kernel(AF af, BF bf, EP ep, int ktiles)
{
    __shared__ __align__(16) bf16_t As[BM][KT + LPAD];   // [m][k]
    __shared__ __align__(16) bf16_t Bs[BN][KT + LPAD];   // [n][k] (transposed)

    const int tid  = threadIdx.x;
    const int lane = tid & 31;
    const int wm   = (tid >> 5) & 3;
    const int wn   = (tid >> 5) >> 2;
    const int row0 = blockIdx.x * BM;
    const int col0 = blockIdx.y * BN;

    // loader mapping: one (row, 16-wide k-chunk) per thread for A and for B
    const int lr = tid & 127;
    const int lc = (tid >> 7) << 4;      // 0 or 16

    v8f acc[2][4];
#pragma unroll
    for (int mi = 0; mi < 2; ++mi)
#pragma unroll
        for (int ni = 0; ni < 4; ++ni)
#pragma unroll
            for (int r = 0; r < 8; ++r) acc[mi][ni][r] = 0.f;

    // ISA fragment geometry (wave32, 16-bit operands):
    //  A: lanes 0-15 -> M=lane, K {0..7,16..23}; lanes 16-31 -> K {8..15,24..31}
    //  B: lanes 0-15 -> N=lane, K {0..7,16..23}; lanes 16-31 -> K {8..15,24..31}
    const int fm   = lane & 15;
    const int kh8  = (lane >> 4) << 3;
    const int kh16 = (lane >> 4) << 4;

    Chunk ca = af.load16(row0 + lr, lc);
    Chunk cb = bf.load16(col0 + lr, lc);

    for (int kt = 0; kt < ktiles; ++kt) {
        __syncthreads();                                   // LDS free
        *(v8bf*)&As[lr][lc]     = ca.lo;
        *(v8bf*)&As[lr][lc + 8] = ca.hi;
        *(v8bf*)&Bs[lr][lc]     = cb.lo;
        *(v8bf*)&Bs[lr][lc + 8] = cb.hi;
        __syncthreads();                                   // LDS filled

        if (kt + 1 < ktiles) {                             // overlap w/ wmma
            const int kb = (kt + 1) * KT + lc;
            ca = af.load16(row0 + lr, kb);
            cb = bf.load16(col0 + lr, kb);
        }

        v16bf afr[2], bfr[4];
#pragma unroll
        for (int mi = 0; mi < 2; ++mi) {
            const bf16_t* ap = &As[wm * 32 + mi * 16 + fm][kh8];
            v8bf lo = *(const v8bf*)ap;
            v8bf hi = *(const v8bf*)(ap + 16);
#pragma unroll
            for (int j = 0; j < 8; ++j) { afr[mi][j] = lo[j]; afr[mi][8 + j] = hi[j]; }
        }
#pragma unroll
        for (int ni = 0; ni < 4; ++ni) {
            const bf16_t* bp = &Bs[wn * 64 + ni * 16 + fm][kh8];
            v8bf lo = *(const v8bf*)bp;
            v8bf hi = *(const v8bf*)(bp + 16);
#pragma unroll
            for (int j = 0; j < 8; ++j) { bfr[ni][j] = lo[j]; bfr[ni][8 + j] = hi[j]; }
        }
#pragma unroll
        for (int mi = 0; mi < 2; ++mi)
#pragma unroll
            for (int ni = 0; ni < 4; ++ni)
                acc[mi][ni] = __builtin_amdgcn_wmma_f32_16x16x32_bf16(
                    false, afr[mi], false, bfr[ni],
                    (short)0, acc[mi][ni], false, false);
    }

    // C/D layout: VGPR r, lanes 0-15 -> M=r, lanes 16-31 -> M=8+r; N=lane&15
    const int rhi = (lane >> 4) << 3;
#pragma unroll
    for (int mi = 0; mi < 2; ++mi)
#pragma unroll
        for (int ni = 0; ni < 4; ++ni)
#pragma unroll
            for (int r = 0; r < 8; ++r)
                ep.store(row0 + wm * 32 + mi * 16 + rhi + r,
                         col0 + wn * 64 + ni * 16 + fm,
                         acc[mi][ni][r]);
}

// ---------------------------------------------------------------------------

static inline size_t align256(size_t x) { return (x + 255) & ~(size_t)255; }

extern "C" void kernel_launch(void* const* d_in, const int* in_sizes, int n_in,
                              void* d_out, int out_size, void* d_ws, size_t ws_size,
                              hipStream_t stream)
{
    const float* n_f_orig = (const float*)d_in[0];
    const float* new_n_f  = (const float*)d_in[1];
    const float* w2v      = (const float*)d_in[2];
    const float* new_lang = (const float*)d_in[3];
    const float* s_f      = (const float*)d_in[4];
    const float* W_node   = (const float*)d_in[5];
    const float* b_node   = (const float*)d_in[6];
    const float* W_lang   = (const float*)d_in[7];
    const float* b_lang   = (const float*)d_in[8];
    const float* W_e1     = (const float*)d_in[9];
    const float* b_e1     = (const float*)d_in[10];
    const float* W_e2     = (const float*)d_in[11];
    const float* b_e2     = (const float*)d_in[12];
    const int*   src_idx  = (const int*)d_in[13];
    const int*   dst_idx  = (const int*)d_in[14];
    float* out = (float*)d_out;

    // Workspace (~268 MB): activations + pre-transposed bf16 weights
    char* ws = (char*)d_ws;
    size_t off = 0;
    bf16_t* nf  = (bf16_t*)(ws + off); off = align256(off + (size_t)N_NODES * D_FEAT * 2);
    bf16_t* nl  = (bf16_t*)(ws + off); off = align256(off + (size_t)N_NODES * NL_LD * 2);
    bf16_t* h   = (bf16_t*)(ws + off); off = align256(off + (size_t)N_EDGES * D_HID * 2);
    bf16_t* wtN = (bf16_t*)(ws + off); off = align256(off + (size_t)NR_NODE * LDK_NODE * 2);
    bf16_t* wtL = (bf16_t*)(ws + off); off = align256(off + (size_t)NR_LANG * LDK_LANG * 2);
    bf16_t* wt1 = (bf16_t*)(ws + off); off = align256(off + (size_t)NR_E1 * LDK_E1 * 2);
    bf16_t* wt2 = (bf16_t*)(ws + off); off = align256(off + (size_t)NR_E2 * LDK_E2 * 2);
    (void)ws_size; (void)in_sizes; (void)n_in; (void)out_size;

    // 0) one-time: transpose + bf16-convert all weights (padded, zero-filled)
    {
        long long t;
        t = (long long)NR_NODE * LDK_NODE;
        transpose_w_kernel<<<dim3((unsigned)((t + 255) / 256)), 256, 0, stream>>>(
            W_node, wtN, 2 * D_FEAT, D_FEAT, LDK_NODE, t);
        t = (long long)NR_LANG * LDK_LANG;
        transpose_w_kernel<<<dim3((unsigned)((t + 255) / 256)), 256, 0, stream>>>(
            W_lang, wtL, 2 * D_LANG, D_LANG, LDK_LANG, t);
        t = (long long)NR_E1 * LDK_E1;
        transpose_w_kernel<<<dim3((unsigned)((t + 255) / 256)), 256, 0, stream>>>(
            W_e1, wt1, K_EDGE, D_HID, LDK_E1, t);
        t = (long long)NR_E2 * LDK_E2;
        transpose_w_kernel<<<dim3((unsigned)((t + 255) / 256)), 256, 0, stream>>>(
            W_e2, wt2, D_HID, N_CLS, LDK_E2, t);
    }

    // 1) nf = relu([n_f_orig | new_n_f] @ W_node + b_node)   [20000 x 1024]
    {
        ConcatA     a{n_f_orig, new_n_f, N_NODES, D_FEAT, 2 * D_FEAT};
        TransB      b{wtN, LDK_NODE};
        ReluBf16Out e{nf, b_node, N_NODES, D_FEAT, D_FEAT};
        dim3 g((N_NODES + BM - 1) / BM, NR_NODE / BN);
        wmma_gemm_kernel<<<g, 256, 0, stream>>>(a, b, e, LDK_NODE / KT);
    }
    // 2) nl = relu([w2v | new_lang] @ W_lang + b_lang)       [20000 x 300->304]
    {
        ConcatA     a{w2v, new_lang, N_NODES, D_LANG, 2 * D_LANG};
        TransB      b{wtL, LDK_LANG};
        ReluBf16Out e{nl, b_lang, N_NODES, D_LANG, NL_LD};
        dim3 g((N_NODES + BM - 1) / BM, NR_LANG / BN);
        wmma_gemm_kernel<<<g, 256, 0, stream>>>(a, b, e, LDK_LANG / KT);
    }
    // 3) h = relu(gather-concat(ef) @ W_e1 + b_e1)           [100000 x 1024]
    {
        EdgeA       a{nf, nl, s_f, src_idx, dst_idx, N_EDGES};
        TransB      b{wt1, LDK_E1};
        ReluBf16Out e{h, b_e1, N_EDGES, D_HID, D_HID};
        dim3 g((N_EDGES + BM - 1) / BM, NR_E1 / BN);
        wmma_gemm_kernel<<<g, 256, 0, stream>>>(a, b, e, LDK_E1 / KT);
    }
    // 4) pred = h @ W_e2 + b_e2                              [100000 x 117]
    {
        Bf16A      a{h, N_EDGES, D_HID, D_HID};
        TransB     b{wt2, LDK_E2};
        BiasF32Out e{out, b_e2, N_EDGES, N_CLS};
        dim3 g((N_EDGES + BM - 1) / BM, NR_E2 / BN);
        wmma_gemm_kernel<<<g, 256, 0, stream>>>(a, b, e, LDK_E2 / KT);
    }
}
